// GRUD_11622181503410
// MI455X (gfx1250) — matrix operand order repaired and checked by reading, not compiled, not run on hardware
//
#include <hip/hip_runtime.h>
#include <hip/hip_bf16.h>
#include <cmath>

// Problem dims (fixed by the reference)
#define Bsz 256
#define Tsz 256
#define Dsz 128
#define Hsz 256
#define Ksz 512   // D + H + D

typedef __attribute__((ext_vector_type(16))) __bf16       v16bf;
typedef __attribute__((ext_vector_type(8)))  float        v8f;
typedef __attribute__((ext_vector_type(4)))  unsigned int v4u;

union FragBF { v4u q[2]; v16bf v; };

__device__ __forceinline__ unsigned short f2bf(float f) {
  unsigned u = __float_as_uint(f);
  u = (u + 0x7FFFu + ((u >> 16) & 1u)) >> 16;   // round-to-nearest-even
  return (unsigned short)u;
}

__device__ __forceinline__ v8f vzero8() {
  v8f z = {0.f, 0.f, 0.f, 0.f, 0.f, 0.f, 0.f, 0.f};
  return z;
}

__device__ __forceinline__ v8f wmma_bf16(v16bf a, v16bf b, v8f c) {
  // (neg_a, A, neg_b, B, c_mod, C, reuse_a, reuse_b)
  return __builtin_amdgcn_wmma_f32_16x16x32_bf16(false, a, false, b, (short)0, c,
                                                 false, false);
}

// A-fragment (16x32 bf16) from a row-major LDS buffer [16][ldk].
// ISA 7.12.2 16-bit A layout: lanes 0-15 hold M=lane, K = {0..7,16..23};
// lanes 16-31 hold M=lane-16, K = {8..15,24..31}.
__device__ __forceinline__ v16bf load_A_lds(const unsigned short* base, int ldk,
                                            int k0, int lane) {
  int row = lane & 15;
  int kb  = k0 + ((lane & 16) ? 8 : 0);
  const unsigned short* p = base + row * ldk + kb;
  FragBF fr;
  fr.q[0] = *(const v4u*)(p);
  fr.q[1] = *(const v4u*)(p + 16);
  return fr.v;
}

// B-fragment (32x16 bf16 = W^T k-slice). W stored [N][ldk] bf16 row-major in
// global, so each lane's 16 K values sit in two contiguous runs of 8 in memory.
// lanes 0-15: N = n0+lane, K = k0+{0..7,16..23} ; lanes 16-31: N = n0+lane-16,
// K = k0+{8..15,24..31}.
__device__ __forceinline__ v16bf load_B_glb(const unsigned short* W, int ldk,
                                            int n0, int k0, int lane) {
  int n  = n0 + (lane & 15);
  int kb = k0 + ((lane & 16) ? 8 : 0);
  const unsigned short* p = W + (size_t)n * ldk + kb;
  FragBF fr;
  fr.q[0] = *(const v4u*)(p);
  fr.q[1] = *(const v4u*)(p + 16);
  return fr.v;
}

// ---------------------------------------------------------------------------
// One-shot weight prep: f32 -> bf16 (WMMA B layout = plain row-major) + diag.
// ---------------------------------------------------------------------------
__global__ void grud_prep(const float* __restrict__ Wz, const float* __restrict__ Wr,
                          const float* __restrict__ Wh, const float* __restrict__ Wgh,
                          const float* __restrict__ Wgx,
                          unsigned short* __restrict__ wz, unsigned short* __restrict__ wr,
                          unsigned short* __restrict__ wh, unsigned short* __restrict__ wgh,
                          float* __restrict__ diag) {
  int i = blockIdx.x * blockDim.x + threadIdx.x;
  if (i < Hsz * Ksz) {
    wz[i] = f2bf(Wz[i]);
    wr[i] = f2bf(Wr[i]);
    wh[i] = f2bf(Wh[i]);
  }
  if (i < Hsz * Dsz) wgh[i] = f2bf(Wgh[i]);
  if (i < Dsz)       diag[i] = Wgx[i * Dsz + i];   // FilterLinear == diagonal
}

// ---------------------------------------------------------------------------
// Main recurrence. Grid = 16 blocks (one per 16-row batch tile), 512 threads =
// 16 waves, wave w owns H columns [16w, 16w+16). Whole T loop runs in-kernel;
// hidden state lives in LDS (f32), GEMM operands staged in LDS as bf16.
// ---------------------------------------------------------------------------
__global__ __launch_bounds__(512, 1) void grud_main(
    const float* __restrict__ X,  const float* __restrict__ Xl,
    const float* __restrict__ Mk, const float* __restrict__ Dl,
    const float* __restrict__ Xm,
    const unsigned short* __restrict__ Wz_bf, const float* __restrict__ bz,
    const unsigned short* __restrict__ Wr_bf, const float* __restrict__ br,
    const unsigned short* __restrict__ Wh_bf, const float* __restrict__ bh,
    const float* __restrict__ wgx_diag, const float* __restrict__ bgx,
    const unsigned short* __restrict__ Wgh_bf, const float* __restrict__ bgh,
    float* __restrict__ out, float* __restrict__ out_last) {
  __shared__ float          hS[16][Hsz];        // carried hidden state (f32)
  __shared__ unsigned short combA[16][Ksz];     // [x | h_pre | m] as bf16
  __shared__ unsigned short rhA[16][Hsz];       // r * h_pre as bf16
  __shared__ unsigned short dA[16][Dsz];        // Delta_t as bf16

  const int tid  = threadIdx.x;
  const int lane = tid & 31;
  const int wave = tid >> 5;        // 0..15
  const int n0   = wave * 16;       // column tile base in H
  const int b0   = blockIdx.x * 16; // batch tile base

  for (int i = tid; i < 16 * Hsz; i += 512) ((float*)hS)[i] = 0.f;

  // Per-lane loop-invariant values (C/D layout: N = lane&15, M = v + hi8)
  const int   ncol = n0 + (lane & 15);
  const int   hi8  = (lane & 16) ? 8 : 0;
  const float bzv  = bz[ncol], brv = br[ncol], bhv = bh[ncol], bghv = bgh[ncol];
  float diagv[4], bgxv[4];
#pragma unroll
  for (int j = 0; j < 4; ++j) {
    int c = (tid + j * 512) & 127;
    diagv[j] = wgx_diag[c];
    bgxv[j]  = bgx[c];
  }

  // Wgh B-fragments are small (4 slices x 8 VGPRs = 32 VGPRs): pin them in
  // registers for the whole recurrence -> phase 2 does no weight loads.
  v16bf wghF[4];
#pragma unroll
  for (int ks = 0; ks < Dsz; ks += 32) wghF[ks / 32] = load_B_glb(Wgh_bf, Dsz, n0, ks, lane);

  // Local copies of the big weight pointers; re-blinded every step so LLVM
  // cannot hoist their (loop-invariant) fragment loads out of the t-loop and
  // spill them to scratch. They stay as in-loop global loads hitting L2.
  const unsigned short* wzp = Wz_bf;
  const unsigned short* wrp = Wr_bf;
  const unsigned short* whp = Wh_bf;

  v8f hpre = vzero8();
  __syncthreads();

  for (int t = 0; t < Tsz; ++t) {
    asm volatile("" : "+s"(wzp));
    asm volatile("" : "+s"(wrp));
    asm volatile("" : "+s"(whp));

    // ---- phase 1: elementwise input decay; stage x_t, m, Delta in LDS ----
#pragma unroll
    for (int j = 0; j < 4; ++j) {
      int e = tid + j * 512;                 // 16*128 = 2048 elements
      int r = e >> 7, c = e & 127;
      size_t g = ((size_t)(b0 + r) * Tsz + t) * Dsz + c;
      float x  = X[g];
      float xl = Xl[g];
      float m  = Mk[g];
      float dl = Dl[g];
      float xm = Xm[t * Dsz + c];
      float dx = __expf(-fmaxf(dl * diagv[j] + bgxv[j], 0.f));
      float xt = m * x + (1.f - m) * (dx * xl + (1.f - dx) * xm);
      combA[r][c]       = f2bf(xt);
      combA[r][384 + c] = f2bf(m);
      dA[r][c]          = f2bf(dl);
      if (t + 1 < Tsz) __builtin_prefetch(&X[g + Dsz], 0, 3);
    }
    __syncthreads();

    // ---- phase 2: delta_h = exp(-relu(Delta @ Wgh^T + bgh)); decay h ----
    v8f acc = vzero8();
#pragma unroll
    for (int ks = 0; ks < Dsz; ks += 32) {
      v16bf a = load_A_lds(&dA[0][0], Dsz, ks, lane);
      acc = wmma_bf16(a, wghF[ks / 32], acc);
    }
#pragma unroll
    for (int v = 0; v < 8; ++v) {
      float dh = __expf(-fmaxf(acc[v] + bghv, 0.f));
      float hp = dh * hS[v + hi8][ncol];     // own columns only: no race
      hpre[v]  = hp;
      combA[v + hi8][Dsz + ncol] = f2bf(hp); // middle section of comb
    }
    __syncthreads();

    // ---- phase 3a: z gate over K=512 ----
    v8f zacc = vzero8();
#pragma unroll
    for (int ks = 0; ks < Ksz; ks += 32) {
      v16bf a  = load_A_lds(&combA[0][0], Ksz, ks, lane);
      v16bf wz = load_B_glb(wzp, Ksz, n0, ks, lane);
      zacc = wmma_bf16(a, wz, zacc);
    }
    v8f zg = vzero8();
#pragma unroll
    for (int v = 0; v < 8; ++v)
      zg[v] = 1.f / (1.f + __expf(-(zacc[v] + bzv)));

    // ---- phase 3b: r gate over K=512 ----
    v8f racc = vzero8();
#pragma unroll
    for (int ks = 0; ks < Ksz; ks += 32) {
      v16bf a  = load_A_lds(&combA[0][0], Ksz, ks, lane);
      v16bf wr = load_B_glb(wrp, Ksz, n0, ks, lane);
      racc = wmma_bf16(a, wr, racc);
    }
#pragma unroll
    for (int v = 0; v < 8; ++v) {
      float rg = 1.f / (1.f + __expf(-(racc[v] + brv)));
      rhA[v + hi8][ncol] = f2bf(rg * hpre[v]);
    }
    __syncthreads();

    // ---- phase 4: h_tilde over [x | r*h | m]; blend; write back ----
    v8f hacc = vzero8();
#pragma unroll
    for (int ks = 0; ks < Ksz; ks += 32) {
      const unsigned short* ab;
      int ldk, kl;
      if (ks < Dsz)            { ab = &combA[0][0]; ldk = Ksz; kl = ks; }
      else if (ks < Dsz + Hsz) { ab = &rhA[0][0];   ldk = Hsz; kl = ks - Dsz; }
      else                     { ab = &combA[0][0]; ldk = Ksz; kl = ks; }
      v16bf a = load_A_lds(ab, ldk, kl, lane);
      v16bf b = load_B_glb(whp, Ksz, n0, ks, lane);
      hacc = wmma_bf16(a, b, hacc);
    }
#pragma unroll
    for (int v = 0; v < 8; ++v) {
      float ht = tanhf(hacc[v] + bhv);
      float hn = (1.f - zg[v]) * hpre[v] + zg[v] * ht;
      hS[v + hi8][ncol] = hn;                         // own columns only
      int brow = b0 + v + hi8;
      out[((size_t)brow * Tsz + t) * Hsz + ncol] = hn;
      if (t == Tsz - 1) out_last[(size_t)brow * Hsz + ncol] = hn;
    }
    __syncthreads();   // protect combA/rhA before next step's phase 1
  }
}

extern "C" void kernel_launch(void* const* d_in, const int* in_sizes, int n_in,
                              void* d_out, int out_size, void* d_ws, size_t ws_size,
                              hipStream_t stream) {
  (void)in_sizes; (void)n_in; (void)out_size; (void)ws_size;

  const float* X   = (const float*)d_in[0];
  const float* Xl  = (const float*)d_in[1];
  const float* Mk  = (const float*)d_in[2];
  const float* Dl  = (const float*)d_in[3];
  const float* Xm  = (const float*)d_in[4];
  const float* Wz  = (const float*)d_in[5];
  const float* bz  = (const float*)d_in[6];
  const float* Wr  = (const float*)d_in[7];
  const float* br  = (const float*)d_in[8];
  const float* Wh  = (const float*)d_in[9];
  const float* bh  = (const float*)d_in[10];
  const float* Wgx = (const float*)d_in[11];
  const float* bgx = (const float*)d_in[12];
  const float* Wgh = (const float*)d_in[13];
  const float* bgh = (const float*)d_in[14];

  // Workspace carve: bf16 weights + diag(Wgx)  (~852 KB)
  unsigned short* wz  = (unsigned short*)d_ws;
  unsigned short* wr  = wz + Hsz * Ksz;
  unsigned short* wh  = wr + Hsz * Ksz;
  unsigned short* wgh = wh + Hsz * Ksz;
  float*          diag = (float*)(wgh + Hsz * Dsz);

  float* out      = (float*)d_out;
  float* out_last = out + (size_t)Bsz * Tsz * Hsz;

  grud_prep<<<dim3((Hsz * Ksz + 255) / 256), dim3(256), 0, stream>>>(
      Wz, Wr, Wh, Wgh, Wgx, wz, wr, wh, wgh, diag);

  grud_main<<<dim3(Bsz / 16), dim3(512), 0, stream>>>(
      X, Xl, Mk, Dl, Xm, wz, bz, wr, br, wh, bh, diag, bgx, wgh, bgh,
      out, out_last);
}
